// SpectralWiseMSA_69174743269724
// MI455X (gfx1250) — compile-verified, matrix operand order checked by reading, not run on hardware
//
#include <hip/hip_runtime.h>
#include <hip/hip_bf16.h>

#define Bn 8
#define Cc 256
#define Hh 128
#define Ww 128
#define HW 16384
#define NHEADS 8
#define DHd 32
#define STR 18    // dwords per 32-elem bf16 LDS row (16 data + 2 pad)
#define WROW 130  // dwords per 256-elem bf16 weight row (128 data + 2 pad)

typedef __attribute__((ext_vector_type(16))) __bf16 v16bf;
typedef __attribute__((ext_vector_type(8)))  float  v8f;

union FragU { v16bf v; unsigned int u[8]; };

// Hardware f32->bf16 convert (RNE).
__device__ __forceinline__ unsigned short f2bf(float f) {
  union { __bf16 h; unsigned short s; } r;
  r.h = (__bf16)f;
  return r.s;
}
// Exactly one v_cvt_pk_bf16_f32: dst.lo = bf16(a), dst.hi = bf16(b).
__device__ __forceinline__ unsigned pack2(float a, float b) {
  unsigned r;
  asm("v_cvt_pk_bf16_f32 %0, %1, %2" : "=v"(r) : "v"(a), "v"(b));
  return r;
}

// A/B fragment from an LDS row of 32 bf16 (16 consecutive dwords).
// lane<16: K = {0..7,16..23}; lane>=16: K = {8..15,24..31}  (ISA 7.12.2)
__device__ __forceinline__ v16bf lds_frag(const unsigned* row, int hi) {
  FragU f;
  const unsigned* p = row + hi * 4;
#pragma unroll
  for (int j = 0; j < 4; ++j) f.u[j] = p[j];
#pragma unroll
  for (int j = 0; j < 4; ++j) f.u[4 + j] = p[8 + j];
  return f.v;
}

// Same fragment pattern straight from a global bf16 row (two 16B loads).
__device__ __forceinline__ v16bf glb_frag(const unsigned short* rowp, int k, int hi) {
  FragU f;
  uint4 qa = *(const uint4*)(rowp + k + hi * 8);
  uint4 qb = *(const uint4*)(rowp + k + 16 + hi * 8);
  f.u[0] = qa.x; f.u[1] = qa.y; f.u[2] = qa.z; f.u[3] = qa.w;
  f.u[4] = qb.x; f.u[5] = qb.y; f.u[6] = qb.z; f.u[7] = qb.w;
  return f.v;
}

__device__ __forceinline__ v8f wmma_bf16(v16bf a, v16bf b, v8f c) {
  return __builtin_amdgcn_wmma_f32_16x16x32_bf16(false, a, false, b, (short)0, c, false, false);
}

// ---------------------------------------------------------------------------
// Kernel 1: Q/K/V projections.  Workgroup: 128 n x 64 co (8 waves), with the
// full 64x256 bf16 tiles of Wq/Wk/Wv persisted in LDS (~100 KB, uses the
// 320 KB WGP LDS) and 4 n-subtiles streamed against them.
// Writes V fp32 (b,c,n) [= (B,C,H,W) for the conv path], Qt/Kt bf16 (b,c,n).
// ---------------------------------------------------------------------------
__global__ __launch_bounds__(256) void qkv_kernel(
    const float* __restrict__ x, const float* __restrict__ Wq,
    const float* __restrict__ Wk, const float* __restrict__ Wv,
    float* __restrict__ V, unsigned short* __restrict__ Qt,
    unsigned short* __restrict__ Kt) {
  __shared__ unsigned sW[3][64 * WROW];  // 99,840 B persistent weights
  __shared__ unsigned sX[32 * STR];
  __shared__ float sD[64 * 33];

  const int t = threadIdx.x;
  const int lane = t & 31, wave = t >> 5;
  const int hi = lane >> 4, lm = lane & 15;
  const int nblk = blockIdx.x * 128;
  const int co0 = blockIdx.y * 64;
  const int bb = blockIdx.z;
  const int wn = (wave >> 2) * 16;  // 0 / 16
  const int wc = (wave & 3) * 16;   // 0..48

  // Stage all three weight matrices once (rows co0..co0+63, full K=256).
  {
    int r = t >> 2, c8 = (t & 3) * 8;
    for (int ci0 = 0; ci0 < Cc; ci0 += 32) {
      size_t off = (size_t)(co0 + r) * Cc + ci0 + c8;
      int dof = r * WROW + (ci0 + c8) / 2;
#pragma unroll
      for (int m = 0; m < 3; ++m) {
        const float* Wm = (m == 0) ? Wq : (m == 1) ? Wk : Wv;
        float4 a0 = *(const float4*)(Wm + off);
        float4 a1 = *(const float4*)(Wm + off + 4);
        unsigned* d = &sW[m][dof];
        d[0] = pack2(a0.x, a0.y); d[1] = pack2(a0.z, a0.w);
        d[2] = pack2(a1.x, a1.y); d[3] = pack2(a1.z, a1.w);
      }
    }
  }
  __syncthreads();

  for (int ns = 0; ns < 4; ++ns) {
    const int n0 = nblk + ns * 32;
    v8f accQ = {}; v8f accK = {}; v8f accV = {};

    for (int ci0 = 0; ci0 < Cc; ci0 += 32) {
      __syncthreads();
      {  // stage X tile (transpose (ci,n) -> LDS [n][ci] bf16)
        int ci = t >> 3, n4 = (t & 7) * 4;
        const float4 v = *(const float4*)(x + (size_t)(bb * Cc + ci0 + ci) * HW + n0 + n4);
        unsigned short* sx = (unsigned short*)sX;
        sx[(n4 + 0) * (2 * STR) + ci] = f2bf(v.x);
        sx[(n4 + 1) * (2 * STR) + ci] = f2bf(v.y);
        sx[(n4 + 2) * (2 * STR) + ci] = f2bf(v.z);
        sx[(n4 + 3) * (2 * STR) + ci] = f2bf(v.w);
      }
      __syncthreads();
      const int wof = (wc + lm) * WROW + ci0 / 2;
      v16bf fa = lds_frag(sX + (wn + lm) * STR, hi);
      v16bf fq = lds_frag(&sW[0][wof], hi);
      v16bf fk = lds_frag(&sW[1][wof], hi);
      v16bf fv = lds_frag(&sW[2][wof], hi);
      accQ = wmma_bf16(fa, fq, accQ);
      accK = wmma_bf16(fa, fk, accK);
      accV = wmma_bf16(fa, fv, accV);
    }

    // Epilogue: LDS transpose (n,co)->(co,n); bf16 (Q,K) / fp32 (V).
    auto stage_write_bf = [&](v8f acc, unsigned short* out) {
      __syncthreads();
      const float* a = (const float*)&acc;
#pragma unroll
      for (int j = 0; j < 8; ++j) sD[(wc + lm) * 33 + wn + j + 8 * hi] = a[j];
      __syncthreads();
      int co = t >> 2, n8 = (t & 3) * 8;
      const float* r = sD + co * 33 + n8;
      uint4 q = make_uint4(pack2(r[0], r[1]), pack2(r[2], r[3]),
                           pack2(r[4], r[5]), pack2(r[6], r[7]));
      *(uint4*)(out + (size_t)(bb * Cc + co0 + co) * HW + n0 + n8) = q;
    };
    stage_write_bf(accQ, Qt);
    stage_write_bf(accK, Kt);
    __syncthreads();
    {
      const float* a = (const float*)&accV;
#pragma unroll
      for (int j = 0; j < 8; ++j) sD[(wc + lm) * 33 + wn + j + 8 * hi] = a[j];
    }
    __syncthreads();
    {
      int co = t >> 2, n8 = (t & 3) * 8;
      const float* r = sD + co * 33 + n8;
      float* p = V + (size_t)(bb * Cc + co0 + co) * HW + n0 + n8;
      *(float4*)p = make_float4(r[0], r[1], r[2], r[3]);
      *(float4*)(p + 4) = make_float4(r[4], r[5], r[6], r[7]);
    }
  }
}

// ---------------------------------------------------------------------------
// Kernel 2: logits[d,e] = sum_n K[n,d]*Q[n,e], softmax(sigma*logits) over e.
// One workgroup per (b,head); 4 waves own the four 16x16 tiles of 32x32.
// ---------------------------------------------------------------------------
__global__ __launch_bounds__(128) void logits_softmax_kernel(
    const unsigned short* __restrict__ Kt, const unsigned short* __restrict__ Qt,
    const float* __restrict__ sigmas, float* __restrict__ attn,
    float* __restrict__ Pspe) {
  __shared__ float sL[32 * 33];
  const int t = threadIdx.x;
  const int lane = t & 31, wave = t >> 5;
  const int hi = lane >> 4, lm = lane & 15;
  const int bb = blockIdx.x >> 3, h = blockIdx.x & 7;
  const int d0 = (wave >> 1) * 16, e0 = (wave & 1) * 16;

  const unsigned short* krow = Kt + (size_t)(bb * Cc + h * DHd + d0 + lm) * HW;
  const unsigned short* qrow = Qt + (size_t)(bb * Cc + h * DHd + e0 + lm) * HW;

  v8f acc = {};
  for (int n = 0; n < HW; n += 32) {
    if ((n & 63) == 0 && n + 1024 < HW) {  // per-lane row prefetch 2KB ahead
      __builtin_prefetch(krow + n + 1024, 0, 0);
      __builtin_prefetch(qrow + n + 1024, 0, 0);
    }
    v16bf fa = glb_frag(krow, n, hi);
    v16bf fb = glb_frag(qrow, n, hi);
    acc = wmma_bf16(fa, fb, acc);
  }
  const float* a = (const float*)&acc;
#pragma unroll
  for (int j = 0; j < 8; ++j) sL[(d0 + j + 8 * hi) * 33 + e0 + lm] = a[j];
  __syncthreads();
  if (t < 32) {
    float sg = sigmas[h];
    float row[32];
    float m = -3.4e38f;
#pragma unroll
    for (int e = 0; e < 32; ++e) { row[e] = sg * sL[t * 33 + e]; m = fmaxf(m, row[e]); }
    float s = 0.f;
#pragma unroll
    for (int e = 0; e < 32; ++e) { row[e] = __expf(row[e] - m); s += row[e]; }
    float inv = 1.f / s;
    float* arow = attn + ((size_t)(bb * NHEADS + h) * DHd + t) * DHd;
    float* prow = Pspe + (size_t)(bb * Cc + h * DHd + t) * Cc + h * DHd;
#pragma unroll
    for (int e = 0; e < 32; ++e) { float v = row[e] * inv; arow[e] = v; prow[e] = v; }
  }
}

// ---------------------------------------------------------------------------
// Kernel 3: head[n,e] = sum_d V_sp[d,n] * attn[d,e]  (per b,head)
// A = V^T tile (M=n, K=d), B = attn (K=d, N=e).  Output written in
// (b, n, c) layout so the proj kernel can fragment-load rows directly.
// ---------------------------------------------------------------------------
__global__ __launch_bounds__(256) void head_kernel(
    const float* __restrict__ V, const float* __restrict__ attn,
    unsigned short* __restrict__ headN) {
  __shared__ unsigned sAt[32 * STR];  // attnT[e][d]  (column e of attn)
  __shared__ unsigned sVt[64 * STR];  // VtT[n][d]
  const int t = threadIdx.x;
  const int lane = t & 31, wave = t >> 5;
  const int hi = lane >> 4, lm = lane & 15;
  const int n0 = blockIdx.x * 64;
  const int h = blockIdx.y, bb = blockIdx.z;
  const int we = (wave & 1) * 16, wn = (wave >> 1) * 16;

  {  // stage attn^T into LDS bf16
    int e = t >> 3, d4 = (t & 7) * 4;
    const float* ap = attn + (size_t)(bb * NHEADS + h) * DHd * DHd;
    unsigned* d = sAt + e * STR + d4 / 2;
    d[0] = pack2(ap[(d4 + 0) * 32 + e], ap[(d4 + 1) * 32 + e]);
    d[1] = pack2(ap[(d4 + 2) * 32 + e], ap[(d4 + 3) * 32 + e]);
  }
  {  // stage V rows (d over n-chunk) transposed into LDS [n][d] bf16
    int d = t >> 3, n8 = (t & 7) * 8;
    const float* vp = V + (size_t)(bb * Cc + h * DHd + d) * HW + n0 + n8;
    float4 a0 = *(const float4*)vp;
    float4 a1 = *(const float4*)(vp + 4);
    unsigned short* sb = (unsigned short*)sVt;
    sb[(n8 + 0) * (2 * STR) + d] = f2bf(a0.x);
    sb[(n8 + 1) * (2 * STR) + d] = f2bf(a0.y);
    sb[(n8 + 2) * (2 * STR) + d] = f2bf(a0.z);
    sb[(n8 + 3) * (2 * STR) + d] = f2bf(a0.w);
    sb[(n8 + 4) * (2 * STR) + d] = f2bf(a1.x);
    sb[(n8 + 5) * (2 * STR) + d] = f2bf(a1.y);
    sb[(n8 + 6) * (2 * STR) + d] = f2bf(a1.z);
    sb[(n8 + 7) * (2 * STR) + d] = f2bf(a1.w);
  }
  __syncthreads();
  v16bf fa = lds_frag(sVt + (wn + lm) * STR, hi);  // A: M=n rows
  v16bf fb = lds_frag(sAt + (we + lm) * STR, hi);  // B: N=e columns
  v8f acc = {};
  acc = wmma_bf16(fa, fb, acc);
  // D: M = n (j+8*hi), N = e (lm); store (b,n,c) row-contiguous.
  const float* a = (const float*)&acc;
#pragma unroll
  for (int j = 0; j < 8; ++j) {
    int n = n0 + wn + j + 8 * hi;
    headN[((size_t)bb * HW + n) * Cc + h * DHd + we + lm] = f2bf(a[j]);
  }
}

// ---------------------------------------------------------------------------
// Kernel 4: X_out[co,n] = sum_c Wo[co,c]*head[n,c] + pos[co,n]  (fp32 out)
// B fragments come straight from global rows of head (b,n,c): no staging.
// ---------------------------------------------------------------------------
__global__ __launch_bounds__(256) void proj_kernel(
    const unsigned short* __restrict__ headN, const float* __restrict__ Wo,
    const float* __restrict__ pos, float* __restrict__ Xout) {
  __shared__ unsigned sA[32 * STR];  // Wo[co][c]
  const int t = threadIdx.x;
  const int lane = t & 31, wave = t >> 5;
  const int hi = lane >> 4, lm = lane & 15;
  const int n0 = blockIdx.x * 64;
  const int co0 = blockIdx.y * 32;
  const int bb = blockIdx.z;
  const int wc = (wave & 1) * 16, wn = (wave >> 1) * 16;

  const unsigned short* brow = headN + ((size_t)bb * HW + n0 + wn + lm) * Cc;

  v8f acc = {};
  for (int c0 = 0; c0 < Cc; c0 += 32) {
    __syncthreads();
    {
      int r = t >> 3, c4 = (t & 7) * 4;
      float4 a = *(const float4*)(Wo + (size_t)(co0 + r) * Cc + c0 + c4);
      unsigned* d = sA + r * STR + c4 / 2;
      d[0] = pack2(a.x, a.y); d[1] = pack2(a.z, a.w);
    }
    __syncthreads();
    v16bf fa = lds_frag(sA + (wc + lm) * STR, hi);
    v16bf fb = glb_frag(brow, c0, hi);
    acc = wmma_bf16(fa, fb, acc);
  }
  const float* a = (const float*)&acc;
#pragma unroll
  for (int j = 0; j < 8; ++j) {
    size_t idx = (size_t)(bb * Cc + co0 + wc + j + 8 * hi) * HW + n0 + wn + lm;
    Xout[idx] = a[j] + pos[idx];
  }
}

// ---------------------------------------------------------------------------
// Depthwise 3x3 conv (SAME), optional exact GELU.  Pure bandwidth kernel.
// ---------------------------------------------------------------------------
__global__ __launch_bounds__(256) void dwconv_kernel(
    const float* __restrict__ in, const float* __restrict__ w,
    const float* __restrict__ bias, float* __restrict__ out, int apply_gelu) {
  size_t idx = (size_t)blockIdx.x * 256 + threadIdx.x;
  int iw = (int)(idx & (Ww - 1));
  int ih = (int)((idx >> 7) & (Hh - 1));
  int c = (int)((idx >> 14) & (Cc - 1));
  const float* wp = w + c * 9;
  const float* base = in + (idx & ~(size_t)(HW - 1));
  float acc = bias[c];
#pragma unroll
  for (int kh = 0; kh < 3; ++kh) {
    int y = ih + kh - 1;
    if ((unsigned)y >= (unsigned)Hh) continue;
#pragma unroll
    for (int kw = 0; kw < 3; ++kw) {
      int xx = iw + kw - 1;
      if ((unsigned)xx >= (unsigned)Ww) continue;
      acc += base[y * Ww + xx] * wp[kh * 3 + kw];
    }
  }
  if (apply_gelu) acc = 0.5f * acc * (1.0f + erff(acc * 0.70710678118f));
  out[idx] = acc;
}

__global__ __launch_bounds__(256) void zero_kernel(float* p, size_t n) {
  size_t i = (size_t)blockIdx.x * 256 + threadIdx.x;
  if (i < n) p[i] = 0.f;
}

extern "C" void kernel_launch(void* const* d_in, const int* in_sizes, int n_in,
                              void* d_out, int out_size, void* d_ws, size_t ws_size,
                              hipStream_t stream) {
  const float* x = (const float*)d_in[0];
  const float* Wq = (const float*)d_in[1];
  const float* Wk = (const float*)d_in[2];
  const float* Wv = (const float*)d_in[3];
  const float* Wo = (const float*)d_in[4];
  const float* sigmas = (const float*)d_in[5];
  const float* dw1_w = (const float*)d_in[6];
  const float* dw1_b = (const float*)d_in[7];
  const float* dw2_w = (const float*)d_in[8];
  const float* dw2_b = (const float*)d_in[9];

  float* Xout = (float*)d_out;
  float* Pspe = Xout + (size_t)Bn * Cc * HW;

  char* ws = (char*)d_ws;
  float* V = (float*)ws;                                              // 128 MB fp32 (b,c,n)
  unsigned short* Qt = (unsigned short*)(ws + (size_t)134217728);     // 64 MB bf16
  unsigned short* Kt = (unsigned short*)(ws + (size_t)201326592);     // 64 MB bf16
  float* tbuf = (float*)(ws + (size_t)134217728);                     // reuses Qt+Kt after logits
  float* pos = (float*)(ws + (size_t)268435456);                      // 128 MB fp32
  unsigned short* headN = (unsigned short*)(ws + (size_t)402653184);  // 64 MB bf16 (b,n,c)
  float* attn = (float*)(ws + (size_t)469762048);                     // 256 KB fp32

  const size_t pspe_n = (size_t)Bn * Cc * Cc;
  zero_kernel<<<(unsigned)((pspe_n + 255) / 256), 256, 0, stream>>>(Pspe, pspe_n);

  qkv_kernel<<<dim3(HW / 128, Cc / 64, Bn), 256, 0, stream>>>(x, Wq, Wk, Wv, V, Qt, Kt);

  logits_softmax_kernel<<<Bn * NHEADS, 128, 0, stream>>>(Kt, Qt, sigmas, attn, Pspe);

  const unsigned nconv = (unsigned)(((size_t)Bn * Cc * HW) / 256);
  dwconv_kernel<<<nconv, 256, 0, stream>>>(V, dw1_w, dw1_b, tbuf, 1);
  dwconv_kernel<<<nconv, 256, 0, stream>>>(tbuf, dw2_w, dw2_b, pos, 0);

  head_kernel<<<dim3(HW / 64, NHEADS, Bn), 256, 0, stream>>>(V, attn, headN);

  proj_kernel<<<dim3(HW / 64, Cc / 32, Bn), 256, 0, stream>>>(headN, Wo, pos, Xout);
}
